// GCN_44246753084090
// MI455X (gfx1250) — matrix operand order repaired and checked by reading, not compiled, not run on hardware
//
#include <hip/hip_runtime.h>

// ---------------------------------------------------------------------------
// CDNA5 (gfx1250, wave32) GCN forward:
//   sup1 = X @ W1            (bf16 WMMA 16x16x32, fp32 accum)
//   h    = relu(SpMM(sup1) + b1)   (edge-parallel fp32 atomics, L2-resident)
//   z    = h @ W2            (bf16 WMMA)
//   out  = log_softmax(SpMM(z) + b2)
// ---------------------------------------------------------------------------

typedef __attribute__((ext_vector_type(16))) __bf16 v16bf;
typedef __attribute__((ext_vector_type(2)))  __bf16 v2bf;
typedef __attribute__((ext_vector_type(8)))  float  v8f;

__device__ __forceinline__ __bf16 f2bf_sw(float f) {
    // round-to-nearest-even fp32 -> bf16 (software fallback)
    unsigned u = __builtin_bit_cast(unsigned, f);
    unsigned r = u + 0x7FFFu + ((u >> 16) & 1u);
    unsigned short h = (unsigned short)(r >> 16);
    return __builtin_bit_cast(__bf16, h);
}

// Convert 8 fp32 -> bf16 into dst[base..base+7]; packed HW cvt when available.
__device__ __forceinline__ void cvt8_to(v16bf& dst, int base, v8f a) {
#if __has_builtin(__builtin_amdgcn_cvt_pk_bf16_f32)
#pragma unroll
    for (int i = 0; i < 4; ++i) {
        v2bf p = __builtin_amdgcn_cvt_pk_bf16_f32(a[2 * i], a[2 * i + 1]);
        dst[base + 2 * i]     = p[0];
        dst[base + 2 * i + 1] = p[1];
    }
#else
#pragma unroll
    for (int i = 0; i < 8; ++i) dst[base + i] = f2bf_sw(a[i]);
#endif
}

// Transpose + convert W[K x N] fp32 (row-major) -> Wt[N x K] bf16 (row-major).
__global__ void transpose_to_bf16(const float* __restrict__ W,
                                  __bf16* __restrict__ Wt, int K, int N) {
    int idx = blockIdx.x * blockDim.x + threadIdx.x;
    if (idx >= K * N) return;
    int nn = idx / K;
    int kk = idx - nn * K;
    Wt[idx] = f2bf_sw(W[(long)kk * N + nn]);
}

// C[n x NTOT] = A[n x K] (fp32 -> bf16 on the fly) @ B, where B is provided as
// Bt[NTOT x K] bf16 (row j of Bt = column j of B, contiguous over K).
// One wave per (16-row tile, 64-col strip): 4 WMMA subtiles, 32 acc VGPRs.
template <int K, int NTOT>
__global__ void __launch_bounds__(256)
gemm_bf16_wmma(const float* __restrict__ A, const __bf16* __restrict__ Bt,
               float* __restrict__ C, int n) {
    constexpr int NSTRIPS = NTOT / 64;
    const int lane = threadIdx.x & 31;
    const int wave = blockIdx.x * (blockDim.x >> 5) + (threadIdx.x >> 5);
    const int mt    = wave / NSTRIPS;          // 16-row tile index
    const int strip = wave - mt * NSTRIPS;     // 64-col strip index
    const int m0 = mt * 16;
    if (m0 >= n) return;          // wave-uniform: EXEC stays all-ones for WMMA
    const int n0 = strip * 64;

    const int ml = lane & 15;     // M (A) / N (B,C) lane coordinate
    const int hi = lane >> 4;     // K-half selector

    int arow = m0 + ml;
    if (arow >= n) arow = n - 1;  // branchless clamp; tail stores are predicated
    const float* Ar = A + (long)arow * K;

    v8f zero = {};
    v8f acc0 = zero, acc1 = zero, acc2 = zero, acc3 = zero;

    for (int kb = 0; kb < K; kb += 32) {
        // A fragment: lane (m,hi) holds K = kb+hi*16+0..15
        v8f a0 = *(const v8f*)(Ar + kb + hi * 16);
        v8f a1 = *(const v8f*)(Ar + kb + hi * 16 + 8);
        v16bf afrag;
        cvt8_to(afrag, 0, a0);
        cvt8_to(afrag, 8, a1);
        const __bf16* Bbase = Bt + (long)(n0 + ml) * K + kb + hi * 16;
        // B fragment j: lane holds column n0+j*16+ml, K = kb+hi*16+0..15
        v16bf b0 = *(const v16bf*)(Bbase + (long)(0 * 16) * K);
        acc0 = __builtin_amdgcn_wmma_f32_16x16x32_bf16(false, afrag, false, b0,
                                                       (short)0, acc0, false, false);
        v16bf b1 = *(const v16bf*)(Bbase + (long)(1 * 16) * K);
        acc1 = __builtin_amdgcn_wmma_f32_16x16x32_bf16(false, afrag, false, b1,
                                                       (short)0, acc1, false, false);
        v16bf b2 = *(const v16bf*)(Bbase + (long)(2 * 16) * K);
        acc2 = __builtin_amdgcn_wmma_f32_16x16x32_bf16(false, afrag, false, b2,
                                                       (short)0, acc2, false, false);
        v16bf b3 = *(const v16bf*)(Bbase + (long)(3 * 16) * K);
        acc3 = __builtin_amdgcn_wmma_f32_16x16x32_bf16(false, afrag, false, b3,
                                                       (short)0, acc3, false, false);
    }

    // C/D layout: VGPR r -> row m0 + r + 8*hi, col = n-lane.
    v8f accs[4] = {acc0, acc1, acc2, acc3};
    float* Cb = C + (long)(m0 + 8 * hi) * NTOT + n0 + ml;
    if (m0 + 16 <= n) {
        // Fast path: full tile, straight-line stores (no exec-mask churn).
#pragma unroll
        for (int j = 0; j < 4; ++j)
#pragma unroll
            for (int r = 0; r < 8; ++r)
                Cb[(long)r * NTOT + j * 16] = accs[j][r];
    } else {
#pragma unroll
        for (int j = 0; j < 4; ++j)
#pragma unroll
            for (int r = 0; r < 8; ++r)
                if (m0 + 8 * hi + r < n) Cb[(long)r * NTOT + j * 16] = accs[j][r];
    }
}

// Edge-parallel SpMM with fp32 global atomics. 32 edges per wave batch: each
// lane non-temporally loads one edge's (src,dst,w) (single-use data; keep L2
// for the gather source), then shuffle-broadcasts. Lanes cover F features with
// stride-32 (128B coalesced) gathers + global_atomic_add_f32 scatters.
template <int F>
__global__ void spmm_atomic(const float* __restrict__ sup,
                            const int* __restrict__ src,
                            const int* __restrict__ dst,
                            const float* __restrict__ ew,
                            float* __restrict__ out, int E) {
    const int lane = threadIdx.x & 31;
    const int wpb = blockDim.x >> 5;
    const int wave = blockIdx.x * wpb + (threadIdx.x >> 5);
    const int totalWaves = gridDim.x * wpb;
    for (int eb = wave * 32; eb < E; eb += totalWaves * 32) {
        int e = eb + lane;
        int s = 0, d = 0;
        float w = 0.0f;
        if (e < E) {
            s = __builtin_nontemporal_load(src + e);
            d = __builtin_nontemporal_load(dst + e);
            w = __builtin_nontemporal_load(ew + e);
        }
        int cnt = E - eb;
        if (cnt > 32) cnt = 32;
        for (int i = 0; i < cnt; ++i) {
            int   si = __shfl(s, i, 32);
            int   di = __shfl(d, i, 32);
            float wi = __shfl(w, i, 32);
            const float* sp = sup + (long)si * F;
            float* op = out + (long)di * F;
#pragma unroll
            for (int t = 0; t < F / 32; ++t) {
                int c = lane + 32 * t;
                atomicAdd(op + c, wi * sp[c]);   // global_atomic_add_f32
            }
        }
    }
}

__global__ void bias_relu(float* __restrict__ h, const float* __restrict__ b,
                          long total, int Fmask) {
    long idx = (long)blockIdx.x * blockDim.x + threadIdx.x;
    if (idx >= total) return;
    float v = h[idx] + b[idx & Fmask];
    h[idx] = v > 0.0f ? v : 0.0f;
}

// In-place per-node log_softmax over 64 classes (+bias). One wave per node,
// two classes per lane, wave32 shuffle reductions.
__global__ void logsoftmax64(float* __restrict__ out, const float* __restrict__ b2,
                             int n) {
    const int lane = threadIdx.x & 31;
    int wave = blockIdx.x * (blockDim.x >> 5) + (threadIdx.x >> 5);
    if (wave >= n) return;
    float* row = out + (long)wave * 64;
    float v0 = row[lane]      + b2[lane];
    float v1 = row[lane + 32] + b2[lane + 32];
    float m = fmaxf(v0, v1);
#pragma unroll
    for (int off = 16; off > 0; off >>= 1)
        m = fmaxf(m, __shfl_xor(m, off, 32));
    float s = __expf(v0 - m) + __expf(v1 - m);
#pragma unroll
    for (int off = 16; off > 0; off >>= 1)
        s += __shfl_xor(s, off, 32);
    float lse = m + __logf(s);
    row[lane]      = v0 - lse;
    row[lane + 32] = v1 - lse;
}

extern "C" void kernel_launch(void* const* d_in, const int* in_sizes, int n_in,
                              void* d_out, int out_size, void* d_ws, size_t ws_size,
                              hipStream_t stream) {
    (void)n_in; (void)out_size; (void)ws_size;
    const float* x   = (const float*)d_in[0];
    const float* w1  = (const float*)d_in[1];
    const float* b1  = (const float*)d_in[2];
    const float* w2  = (const float*)d_in[3];
    const float* b2  = (const float*)d_in[4];
    const float* ew  = (const float*)d_in[5];
    const int* esrc  = (const int*)d_in[6];
    const int* edst  = (const int*)d_in[7];

    constexpr int IN_FEAT = 512, HIDDEN = 256, CLASSES = 64;
    const int n = in_sizes[0] / IN_FEAT;
    const int E = in_sizes[5];

    // Workspace layout (z aliases sup1: sup1 is dead once spmm1 finished).
    char* ws = (char*)d_ws;
    size_t off = 0;
    float* sup1 = (float*)(ws + off); off += (size_t)n * HIDDEN * sizeof(float);
    float* h    = (float*)(ws + off); off += (size_t)n * HIDDEN * sizeof(float);
    __bf16* w1t = (__bf16*)(ws + off); off += (size_t)HIDDEN * IN_FEAT * sizeof(__bf16);
    __bf16* w2t = (__bf16*)(ws + off); off += (size_t)CLASSES * HIDDEN * sizeof(__bf16);
    float* z    = sup1;                // alias
    float* outf = (float*)d_out;

    // 1) Weight transpose + bf16 convert (tiny).
    transpose_to_bf16<<<(IN_FEAT * HIDDEN + 255) / 256, 256, 0, stream>>>(
        w1, w1t, IN_FEAT, HIDDEN);
    transpose_to_bf16<<<(HIDDEN * CLASSES + 255) / 256, 256, 0, stream>>>(
        w2, w2t, HIDDEN, CLASSES);

    const int mtiles = (n + 15) / 16;

    // 2) GEMM1: sup1 = x @ W1. waves = mtiles * 4 strips (same-row strips
    //    adjacent in a block -> X loads coalesce in L2/L0).
    {
        int waves = mtiles * (HIDDEN / 64);
        gemm_bf16_wmma<IN_FEAT, HIDDEN><<<(waves + 7) / 8, 256, 0, stream>>>(
            x, w1t, sup1, n);
    }

    // 3) SpMM1 into h (zeroed first, then atomics), then bias+ReLU.
    hipMemsetAsync(h, 0, (size_t)n * HIDDEN * sizeof(float), stream);
    spmm_atomic<HIDDEN><<<8192, 256, 0, stream>>>(sup1, esrc, edst, ew, h, E);
    {
        long total = (long)n * HIDDEN;
        bias_relu<<<(unsigned)((total + 255) / 256), 256, 0, stream>>>(
            h, b1, total, HIDDEN - 1);
    }

    // 4) GEMM2: z = h @ W2 (single 64-wide strip per wave).
    {
        int waves = mtiles * (CLASSES / 64);
        gemm_bf16_wmma<HIDDEN, CLASSES><<<(waves + 7) / 8, 256, 0, stream>>>(
            h, w2t, z, n);
    }

    // 5) SpMM2 into d_out (zeroed first).
    hipMemsetAsync(outf, 0, (size_t)n * CLASSES * sizeof(float), stream);
    spmm_atomic<CLASSES><<<8192, 256, 0, stream>>>(z, esrc, edst, ew, outf, E);

    // 6) In-place bias + log_softmax.
    logsoftmax64<<<(n + 7) / 8, 256, 0, stream>>>(outf, b2, n);
}
